// ProbAttentionLayer_4827543241298
// MI455X (gfx1250) — matrix operand order repaired and checked by reading, not compiled, not run on hardware
//
#include <hip/hip_runtime.h>
#include <hip/hip_bf16.h>
#include <math.h>

#define B_   4
#define L_   2048
#define D_   512
#define H_   8
#define DK_  64
#define NTOP 24
#define M_   (B_ * L_)   // 8192 rows in the big GEMMs

typedef __attribute__((ext_vector_type(16))) __bf16 v16bf;
typedef __attribute__((ext_vector_type(8)))  __bf16 v8bf;
typedef __attribute__((ext_vector_type(8)))  float  v8f;
typedef __attribute__((ext_vector_type(4)))  int    v4i;

// ---------------------------------------------------------------------------
// 16-byte global -> LDS copy. Prefer the CDNA5 async-DMA path (ASYNCcnt);
// fall back to a plain b128 load/store if the builtin isn't declared.
// Builtin's param 0 is a v4i pointer in the global AS (per hipcc diagnostic).
// ---------------------------------------------------------------------------
__device__ __forceinline__ void cp16(void* lds, const void* g)
{
#if __has_builtin(__builtin_amdgcn_global_load_async_to_lds_b128)
    typedef v4i __attribute__((address_space(1)))* gp_t;
    typedef v4i __attribute__((address_space(3)))* lp_t;
    __builtin_amdgcn_global_load_async_to_lds_b128(
        (gp_t)(v4i*)(void*)(size_t)(const char*)g,
        (lp_t)(v4i*)lds,
        0, 0);
#else
    *(v8bf*)lds = *(const v8bf*)g;
#endif
}

__device__ __forceinline__ void wait_async_all()
{
#if __has_builtin(__builtin_amdgcn_s_wait_asynccnt)
    __builtin_amdgcn_s_wait_asynccnt(0);
#endif
}

// ---------------------------------------------------------------------------
// Precision prep: fp32 -> bf16 (vectorized, 8 elems/thread)
// ---------------------------------------------------------------------------
__global__ __launch_bounds__(256)
void cvt_bf16(const float* __restrict__ src, __bf16* __restrict__ dst, int n8)
{
    int id = blockIdx.x * 256 + threadIdx.x;
    if (id >= n8) return;
    float4 a = ((const float4*)src)[id * 2];
    float4 b = ((const float4*)src)[id * 2 + 1];
    v8bf o;
    o[0] = (__bf16)a.x; o[1] = (__bf16)a.y; o[2] = (__bf16)a.z; o[3] = (__bf16)a.w;
    o[4] = (__bf16)b.x; o[5] = (__bf16)b.y; o[6] = (__bf16)b.z; o[7] = (__bf16)b.w;
    *(v8bf*)&dst[(size_t)id * 8] = o;
}

// W (K,N) fp32 -> Wt (N,K) bf16 (pre-transpose so GEMM B-tiles are contiguous-K)
__global__ __launch_bounds__(256)
void wt_bf16(const float* __restrict__ W, __bf16* __restrict__ Wt)
{
    int id = blockIdx.x * 256 + threadIdx.x;    // id = n*D + k
    int n = id >> 9, k = id & (D_ - 1);
    Wt[id] = (__bf16)W[(size_t)k * D_ + n];
}

// ---------------------------------------------------------------------------
// bf16 WMMA GEMM: out = A(MxK) @ Wt(N,K)^T + bias, fp32 accumulate.
// A and Wt pre-converted bf16, both row-major along K -> LDS staging is pure
// 16B chunk copies (async-to-LDS capable). 256 threads = 8 waves; block tile
// 128x64, BK=64 (2 WMMA K-steps/stage), double-buffered LDS.
// MODE 0: out row-major (M,N).  MODE 1: out scattered to (B,H,L,DK).
// ---------------------------------------------------------------------------
template<int MODE>
__global__ __launch_bounds__(256)
void gemm_bf16(const __bf16* __restrict__ A, const __bf16* __restrict__ Wt,
               const float* __restrict__ bias, float* __restrict__ out,
               int M, int N, int K)
{
    constexpr int BM = 128, BN = 64, BK = 64, PAD = 8;
    __shared__ __bf16 As[2][BM][BK + PAD];   // row stride 144B (16B aligned chunks)
    __shared__ __bf16 Bs[2][BN][BK + PAD];

    const int tid   = threadIdx.x;
    const int wave  = tid >> 5;
    const int lane  = tid & 31;
    const int waveM = wave & 3;              // 4 waves along M
    const int waveN = wave >> 2;             // 2 waves along N
    const int blockM = blockIdx.y * BM;
    const int blockN = blockIdx.x * BN;
    const int lrow  = lane & 15;
    const int group = lane >> 4;

    v8f acc[2][2];
    for (int nf = 0; nf < 2; ++nf) {
        float bv = bias[blockN + waveN * 32 + nf * 16 + lrow];
        for (int mf = 0; mf < 2; ++mf)
            for (int r = 0; r < 8; ++r) acc[mf][nf][r] = bv;
    }

    // Stage one BK-deep tile pair into LDS buffer `buf` (all 16B chunks).
    auto stage = [&](int kt, int buf) {
        // A tile 128x64 bf16 = 1024 chunks -> 4/thread
        for (int i = 0; i < 4; ++i) {
            int id = i * 256 + tid;
            int r = id >> 3, c = id & 7;     // 8 chunks per row
            cp16(&As[buf][r][c * 8], A + (size_t)(blockM + r) * K + kt + c * 8);
        }
        // B tile 64x64 bf16 = 512 chunks -> 2/thread
        for (int i = 0; i < 2; ++i) {
            int id = i * 256 + tid;
            int n = id >> 3, c = id & 7;
            cp16(&Bs[buf][n][c * 8], Wt + (size_t)(blockN + n) * K + kt + c * 8);
        }
    };

    stage(0, 0);
    wait_async_all();
    __syncthreads();

    for (int kt = 0; kt < K; kt += BK) {
        const int buf = (kt >> 6) & 1;
        const bool more = (kt + BK) < K;
        if (more) stage(kt + BK, buf ^ 1);   // prefetch while computing

        for (int ks = 0; ks < BK; ks += 32) {
            // A frag: lane l holds row l%16; K = ks + 16*(j>>3) + (l/16)*8 + (j&7)
            v16bf aF[2], bF[2];
            for (int mf = 0; mf < 2; ++mf) {
                int m = waveM * 32 + mf * 16 + lrow;
                v8bf lo = *(const v8bf*)&As[buf][m][ks + group * 8];
                v8bf hi = *(const v8bf*)&As[buf][m][ks + 16 + group * 8];
                for (int j = 0; j < 8; ++j) { aF[mf][j] = lo[j]; aF[mf][8 + j] = hi[j]; }
            }
            // B frag: lane l holds col l%16; K = ks + 16*(j>>3) + (l/16)*8 + (j&7)
            for (int nf = 0; nf < 2; ++nf) {
                int n = waveN * 32 + nf * 16 + lrow;
                v8bf lo = *(const v8bf*)&Bs[buf][n][ks + group * 8];
                v8bf hi = *(const v8bf*)&Bs[buf][n][ks + 16 + group * 8];
                for (int j = 0; j < 8; ++j) { bF[nf][j] = lo[j]; bF[nf][8 + j] = hi[j]; }
            }
            for (int mf = 0; mf < 2; ++mf)
                for (int nf = 0; nf < 2; ++nf)
                    acc[mf][nf] = __builtin_amdgcn_wmma_f32_16x16x32_bf16(
                        false, aF[mf], false, bF[nf], (short)0, acc[mf][nf], false, false);
        }
        __syncthreads();                     // everyone done reading buf
        if (more) { wait_async_all(); __syncthreads(); }   // next buffer is filled
    }

    // Epilogue: c[r] -> row = 8*(l/16)+r, col = l%16 within the 16x16 tile.
    for (int mf = 0; mf < 2; ++mf)
        for (int nf = 0; nf < 2; ++nf)
            for (int r = 0; r < 8; ++r) {
                int row = blockM + waveM * 32 + mf * 16 + group * 8 + r;
                int col = blockN + waveN * 32 + nf * 16 + lrow;
                float v = acc[mf][nf][r];
                if (MODE == 0) {
                    out[(size_t)row * N + col] = v;
                } else {
                    int b = row >> 11, l = row & (L_ - 1);
                    int h = col >> 6,  dk = col & (DK_ - 1);
                    out[(size_t)((b * H_ + h) * L_ + l) * DK_ + dk] = v;
                }
            }
}

// ---------------------------------------------------------------------------
// m[b,h,l] = max_s(q . k_sample_s) - sum_s(.)/L : one wave per (b,h,l)
// ---------------------------------------------------------------------------
__global__ __launch_bounds__(256)
void sample_scores(const float* __restrict__ Qp, const float* __restrict__ Kp,
                   const int* __restrict__ idxs, float* __restrict__ m)
{
    int wave = threadIdx.x >> 5, lane = threadIdx.x & 31;
    int gid = blockIdx.x * 8 + wave;             // (b*H+h)*L + l
    if (gid >= B_ * H_ * L_) return;
    int l  = gid & (L_ - 1);
    int bh = gid >> 11;
    const float* q = Qp + (size_t)gid * DK_;
    float q0 = q[lane], q1 = q[lane + 32];
    float mx = -INFINITY, sm = 0.f;
    for (int s = 0; s < NTOP; ++s) {
        int ki = idxs[l * NTOP + s];
        const float* kr = Kp + ((size_t)bh * L_ + ki) * DK_;
        float p = q0 * kr[lane] + q1 * kr[lane + 32];
        for (int off = 16; off > 0; off >>= 1) p += __shfl_xor(p, off, 32);
        mx = fmaxf(mx, p);
        sm += p;
    }
    if (lane == 0) m[gid] = mx - sm / (float)L_;
}

// ---------------------------------------------------------------------------
// top-24 per (b,h): iterative argmax, ties to lower index (matches lax.top_k)
// ---------------------------------------------------------------------------
__global__ __launch_bounds__(256)
void topk_kernel(const float* __restrict__ m, int* __restrict__ mtop)
{
    __shared__ float buf[L_];
    __shared__ float sval[256];
    __shared__ int   sidx[256];
    int bh = blockIdx.x, tid = threadIdx.x;
    for (int i = tid; i < L_; i += 256) buf[i] = m[(size_t)bh * L_ + i];
    __syncthreads();
    for (int it = 0; it < NTOP; ++it) {
        float bv = -INFINITY; int bi = 0;
        for (int i = tid * 8; i < tid * 8 + 8; ++i) {
            float v = buf[i];
            if (v > bv) { bv = v; bi = i; }
        }
        sval[tid] = bv; sidx[tid] = bi;
        __syncthreads();
        for (int s = 128; s > 0; s >>= 1) {
            if (tid < s) {
                float v2 = sval[tid + s]; int i2 = sidx[tid + s];
                if (v2 > sval[tid] || (v2 == sval[tid] && i2 < sidx[tid])) {
                    sval[tid] = v2; sidx[tid] = i2;
                }
            }
            __syncthreads();
        }
        if (tid == 0) { mtop[bh * NTOP + it] = sidx[0]; buf[sidx[0]] = -INFINITY; }
        __syncthreads();
    }
}

// ---------------------------------------------------------------------------
// Masked softmax(qK^T/sqrt(D)) @ V for the 24 selected rows: block per row.
// ---------------------------------------------------------------------------
__global__ __launch_bounds__(256)
void attn_kernel(const float* __restrict__ Qp, const float* __restrict__ Kp,
                 const float* __restrict__ Vp, const int* __restrict__ mtop,
                 float* __restrict__ attnv)
{
    __shared__ float qs[DK_];
    __shared__ float sc[L_];
    __shared__ float red[256];
    int blk = blockIdx.x;                        // (b*H+h)*NTOP + u
    int bh  = blk / NTOP;
    int tid = threadIdx.x;
    int row = mtop[blk];
    const float* q = Qp + ((size_t)bh * L_ + row) * DK_;
    if (tid < DK_) qs[tid] = q[tid];
    __syncthreads();
    const float scale = 1.f / sqrtf((float)D_);
    float lmax = -INFINITY;
    for (int k = tid; k < L_; k += 256) {
        const float* kr = Kp + ((size_t)bh * L_ + k) * DK_;
        float d = 0.f;
        for (int j = 0; j < DK_; ++j) d += qs[j] * kr[j];
        d *= scale;
        if (k > row) d = -INFINITY;              // causal mask on selected row
        sc[k] = d;
        lmax = fmaxf(lmax, d);
    }
    red[tid] = lmax; __syncthreads();
    for (int s = 128; s > 0; s >>= 1) { if (tid < s) red[tid] = fmaxf(red[tid], red[tid + s]); __syncthreads(); }
    float gmax = red[0];
    __syncthreads();
    float lsum = 0.f;
    for (int k = tid; k < L_; k += 256) {
        float e = expf(sc[k] - gmax);
        sc[k] = e; lsum += e;
    }
    red[tid] = lsum; __syncthreads();
    for (int s = 128; s > 0; s >>= 1) { if (tid < s) red[tid] += red[tid + s]; __syncthreads(); }
    float inv = 1.f / red[0];
    __syncthreads();
    int g = tid >> 6, dk = tid & (DK_ - 1);
    float acc = 0.f;
    for (int k = g; k < L_; k += 4)
        acc += sc[k] * Vp[((size_t)bh * L_ + k) * DK_ + dk];
    red[tid] = acc; __syncthreads();
    if (g == 0) {
        float t = red[dk] + red[64 + dk] + red[128 + dk] + red[192 + dk];
        attnv[(size_t)blk * DK_ + dk] = t * inv;
    }
}

// ---------------------------------------------------------------------------
// context = cumsum(V, axis=L), written directly in (B, L, D) layout.
// ---------------------------------------------------------------------------
__global__ __launch_bounds__(256)
void cumsum_kernel(const float* __restrict__ Vp, float* __restrict__ ctx)
{
    int id = blockIdx.x * 256 + threadIdx.x;     // (b*H+h)*DK + dk
    if (id >= B_ * H_ * DK_) return;
    int dk = id & (DK_ - 1);
    int bh = id >> 6;
    int b = bh >> 3, h = bh & 7;
    float acc = 0.f;
    for (int l = 0; l < L_; ++l) {
        acc += Vp[((size_t)bh * L_ + l) * DK_ + dk];
        ctx[((size_t)(b * L_ + l)) * D_ + h * DK_ + dk] = acc;
    }
}

// ---------------------------------------------------------------------------
// Scatter attention rows over the cumsum context (top-k rows are distinct).
// ---------------------------------------------------------------------------
__global__ __launch_bounds__(256)
void scatter_kernel(const float* __restrict__ attnv, const int* __restrict__ mtop,
                    float* __restrict__ ctx)
{
    int id = blockIdx.x * 256 + threadIdx.x;     // blk*DK + dk
    if (id >= B_ * H_ * NTOP * DK_) return;
    int dk = id & (DK_ - 1);
    int blk = id >> 6;
    int bh = blk / NTOP;
    int b = bh >> 3, h = bh & 7;
    int row = mtop[blk];
    ctx[((size_t)(b * L_ + row)) * D_ + h * DK_ + dk] = attnv[id];
}

// ---------------------------------------------------------------------------
extern "C" void kernel_launch(void* const* d_in, const int* in_sizes, int n_in,
                              void* d_out, int out_size, void* d_ws, size_t ws_size,
                              hipStream_t stream)
{
    const float* queries = (const float*)d_in[0];
    const float* keys    = (const float*)d_in[1];
    const float* values  = (const float*)d_in[2];
    const int*   idxs    = (const int*)  d_in[3];
    const float* Wq = (const float*)d_in[4];
    const float* bq = (const float*)d_in[5];
    const float* Wk = (const float*)d_in[6];
    const float* bk = (const float*)d_in[7];
    const float* Wv = (const float*)d_in[8];
    const float* bv = (const float*)d_in[9];
    const float* Wo = (const float*)d_in[10];
    const float* bo = (const float*)d_in[11];
    float* out = (float*)d_out;

    char* ws = (char*)d_ws;
    size_t off = 0;
    auto alloc = [&](size_t bytes) {
        void* p = ws + off;
        off += (bytes + 255) & ~(size_t)255;
        return p;
    };
    float*  Qp    = (float*) alloc(sizeof(float) * B_ * H_ * L_ * DK_);   // 16 MB
    float*  Kp    = (float*) alloc(sizeof(float) * B_ * H_ * L_ * DK_);   // 16 MB
    float*  Vp    = (float*) alloc(sizeof(float) * B_ * H_ * L_ * DK_);   // 16 MB
    float*  ctx   = (float*) alloc(sizeof(float) * B_ * L_ * D_);         // 16 MB
    __bf16* Xb    = (__bf16*)alloc(sizeof(__bf16) * M_ * D_);             // 8 MB (reused)
    __bf16* Wtb   = (__bf16*)alloc(sizeof(__bf16) * D_ * D_);             // 512 KB (reused)
    float*  mbuf  = (float*) alloc(sizeof(float) * B_ * H_ * L_);         // 256 KB
    int*    mtop  = (int*)   alloc(sizeof(int)   * B_ * H_ * NTOP);
    float*  attnv = (float*) alloc(sizeof(float) * B_ * H_ * NTOP * DK_);
    (void)ws_size; (void)in_sizes; (void)n_in; (void)out_size;

    const int nCvt = (M_ * D_) / 8;              // 8 elems per thread
    dim3 gGemm(D_ / 64, M_ / 128);               // (8, 64)

    // Q/K/V projections: fp32->bf16 once, W pre-transposed, WMMA GEMM.
    cvt_bf16<<<nCvt / 256, 256, 0, stream>>>(queries, Xb, nCvt);
    wt_bf16<<<(D_ * D_) / 256, 256, 0, stream>>>(Wq, Wtb);
    gemm_bf16<1><<<gGemm, 256, 0, stream>>>(Xb, Wtb, bq, Qp, M_, D_, D_);

    cvt_bf16<<<nCvt / 256, 256, 0, stream>>>(keys, Xb, nCvt);
    wt_bf16<<<(D_ * D_) / 256, 256, 0, stream>>>(Wk, Wtb);
    gemm_bf16<1><<<gGemm, 256, 0, stream>>>(Xb, Wtb, bk, Kp, M_, D_, D_);

    cvt_bf16<<<nCvt / 256, 256, 0, stream>>>(values, Xb, nCvt);
    wt_bf16<<<(D_ * D_) / 256, 256, 0, stream>>>(Wv, Wtb);
    gemm_bf16<1><<<gGemm, 256, 0, stream>>>(Xb, Wtb, bv, Vp, M_, D_, D_);

    // Sparsity measure, top-k selection, attention on selected rows.
    sample_scores<<<(B_ * H_ * L_) / 8, 256, 0, stream>>>(Qp, Kp, idxs, mbuf);
    topk_kernel<<<B_ * H_, 256, 0, stream>>>(mbuf, mtop);
    attn_kernel<<<B_ * H_ * NTOP, 256, 0, stream>>>(Qp, Kp, Vp, mtop, attnv);

    // Cumsum context + scatter attention rows, then output projection.
    cumsum_kernel<<<(B_ * H_ * DK_ + 255) / 256, 256, 0, stream>>>(Vp, ctx);
    scatter_kernel<<<(B_ * H_ * NTOP * DK_ + 255) / 256, 256, 0, stream>>>(attnv, mtop, ctx);

    cvt_bf16<<<nCvt / 256, 256, 0, stream>>>(ctx, Xb, nCvt);
    wt_bf16<<<(D_ * D_) / 256, 256, 0, stream>>>(Wo, Wtb);
    gemm_bf16<0><<<gGemm, 256, 0, stream>>>(Xb, Wtb, bo, out, M_, D_, D_);
}
